// ImprovedSpikingModelV1_35510789604044
// MI455X (gfx1250) — matrix operand order repaired and checked by reading, not compiled, not run on hardware
//
#include <hip/hip_runtime.h>
#include <hip/hip_bf16.h>
#include <cstdint>

// ---------------- problem constants ----------------
#define TT 800
#define BB 32
#define DD 440
#define HH 1024
#define LL 4
#define OO 2000
#define MM (TT * BB)          // 25600 rows
#define NN (2 * HH)           // 2048 cols per recurrent GEMM
#define KPAD0 448             // 440 padded to multiple of 32
#define XLD 1024              // leading dim of bf16 activation buffer
#define NPADF 2048            // wf rows padded 2000 -> 2048

typedef __attribute__((ext_vector_type(16))) __bf16 v16bf;
typedef __attribute__((ext_vector_type(8)))  float  v8f;

__device__ __forceinline__ unsigned short f2bf(float f) {
    unsigned u = __float_as_uint(f);
    unsigned r = u + 0x7FFFu + ((u >> 16) & 1u);   // round-to-nearest-even
    return (unsigned short)(r >> 16);
}

// CDNA5 async global->LDS copy (ASYNCcnt-tracked), 16 bytes per lane.
// vdst = per-lane LDS byte offset, vaddr = per-lane 64-bit global address.
__device__ __forceinline__ void issue_async_b128(unsigned lds_off, const void* g) {
    asm volatile("global_load_async_to_lds_b128 %0, %1, off"
                 :: "v"(lds_off), "v"((unsigned long long)(uintptr_t)g)
                 : "memory");
}

// ---------------- fp32 -> bf16 converters ----------------
__global__ __launch_bounds__(256) void convert_x_kernel(const float* __restrict__ xs,
                                                        unsigned short* __restrict__ xbf) {
    int idx = blockIdx.x * 256 + threadIdx.x;          // over MM * KPAD0
    if (idx >= MM * KPAD0) return;
    int row = idx / KPAD0, col = idx - row * KPAD0;
    float v = (col < DD) ? xs[(size_t)row * DD + col] : 0.0f;
    xbf[(size_t)row * XLD + col] = f2bf(v);
}

__global__ __launch_bounds__(256) void convert_w_kernel(const float* __restrict__ W,
                                                        unsigned short* __restrict__ Wb,
                                                        int N, int K, int Kpad, int total) {
    int idx = blockIdx.x * 256 + threadIdx.x;          // over Npad * Kpad
    if (idx >= total) return;
    int n = idx / Kpad, k = idx - n * Kpad;
    float v = (n < N && k < K) ? W[(size_t)n * K + k] : 0.0f;
    Wb[idx] = f2bf(v);
}

// fold BN into y = x*scale + shift
__global__ void bnprep_kernel(const float* __restrict__ g, const float* __restrict__ b,
                              const float* __restrict__ m, const float* __restrict__ v,
                              float* __restrict__ scale, float* __restrict__ shift, int n) {
    int i = blockIdx.x * 256 + threadIdx.x;
    if (i >= n) return;
    float s = g[i] * rsqrtf(v[i] + 1e-5f);
    scale[i] = s;
    shift[i] = b[i] - m[i] * s;
}

// ---------------- WMMA GEMM ----------------
// C[M,N] = A[M,K](bf16,row-major,lda) * Bw[N,K](bf16,row-major,ldb)^T
// Block tile 128(M) x 128(N), 8 waves (2x4), wave tile 64x32 = 8 wmma tiles.
// Double-buffered LDS filled by global_load_async_to_lds_b128 (ASYNCcnt).
// MODE 1: fused BN + sigmoid->Z / relu->C split epilogue.  MODE 0: raw fp32 store.
template <int MODE>
__global__ __launch_bounds__(256)
void gemm_wmma_bf16(const unsigned short* __restrict__ A, int lda, int K,
                    const unsigned short* __restrict__ Bw, int ldb,
                    const float* __restrict__ scale, const float* __restrict__ shift,
                    float* __restrict__ Zb, float* __restrict__ Cb,
                    float* __restrict__ Out, int ldo, int nBlocksN) {
    const int bid = blockIdx.x;
    const int bm = bid / nBlocksN, bn = bid - bm * nBlocksN;
    const int rowBase = bm * 128, colBase = bn * 128;
    const int tid  = threadIdx.x;
    const int lane = tid & 31;
    const int wave = tid >> 5;
    const int wm = wave >> 2;   // 0..1 : 64-row strip
    const int wn = wave & 3;    // 0..3 : 32-col strip

    __shared__ __align__(16) unsigned short sA[2][128 * 32];   // 2 x 8KB
    __shared__ __align__(16) unsigned short sB[2][128 * 32];   // 2 x 8KB

    const unsigned baseA = (unsigned)(uintptr_t)(void*)&sA[0][0];
    const unsigned baseB = (unsigned)(uintptr_t)(void*)&sB[0][0];

    // staging pattern: each thread async-copies 2 b128 chunks of A and 2 of B:
    // rows r and r+64 at column chunk c (8 bf16 = 16B)
    const int r = tid >> 2;
    const int c = (tid & 3) * 8;
    const unsigned ldsElem = (unsigned)(r * 32 + c) * 2u;

    v8f acc[4][2] = {};

    const int kSteps = K / 32;

#define STAGE(s_, buf_)                                                           \
    do {                                                                          \
        const int k0_ = (s_) * 32;                                                \
        const unsigned short* ga_ = A + (size_t)(rowBase + r) * lda + k0_ + c;    \
        const unsigned short* gb_ = Bw + (size_t)(colBase + r) * ldb + k0_ + c;   \
        const unsigned la_ = baseA + (buf_) * (128 * 32 * 2) + ldsElem;           \
        const unsigned lb_ = baseB + (buf_) * (128 * 32 * 2) + ldsElem;           \
        issue_async_b128(la_, ga_);                                               \
        issue_async_b128(la_ + 64 * 32 * 2, ga_ + (size_t)64 * lda);              \
        issue_async_b128(lb_, gb_);                                               \
        issue_async_b128(lb_ + 64 * 32 * 2, gb_ + (size_t)64 * ldb);              \
    } while (0)

    STAGE(0, 0);

    for (int s = 0; s < kSteps; ++s) {
        const int buf = s & 1;
        if (s + 1 < kSteps) {
            STAGE(s + 1, buf ^ 1);
            asm volatile("s_wait_asynccnt 4" ::: "memory");   // previous batch done
        } else {
            asm volatile("s_wait_asynccnt 0" ::: "memory");
        }
        __syncthreads();

        const unsigned short* pA = &sA[buf][0];
        const unsigned short* pB = &sB[buf][0];

        // A fragments (16-bit 16x32): row = lane&15, K-chunks {0..7,16..23}/{8..15,24..31}
        v16bf af[4], bfr[2];
        const int aOff = (lane >> 4) * 8;
        #pragma unroll
        for (int tm = 0; tm < 4; ++tm) {
            const unsigned short* p = pA + (wm * 64 + tm * 16 + (lane & 15)) * 32;
            union { v16bf v; uint4 q[2]; } u;
            u.q[0] = *(const uint4*)(p + aOff);
            u.q[1] = *(const uint4*)(p + 16 + aOff);
            af[tm] = u.v;
        }
        // B fragments: col = lane&15, K-chunks {0..7,16..23}/{8..15,24..31}
        const int bOff = (lane >> 4) * 8;
        #pragma unroll
        for (int tn = 0; tn < 2; ++tn) {
            const unsigned short* p = pB + (wn * 32 + tn * 16 + (lane & 15)) * 32 + bOff;
            union { v16bf v; uint4 q[2]; } u;
            u.q[0] = *(const uint4*)(p);
            u.q[1] = *(const uint4*)(p + 16);
            bfr[tn] = u.v;
        }
        #pragma unroll
        for (int tm = 0; tm < 4; ++tm)
            #pragma unroll
            for (int tn = 0; tn < 2; ++tn)
                acc[tm][tn] = __builtin_amdgcn_wmma_f32_16x16x32_bf16(
                    false, af[tm], false, bfr[tn], (short)0, acc[tm][tn], false, false);

        __syncthreads();
    }
#undef STAGE

    // epilogue: C/D layout m = v + 8*(lane>>4), n = lane&15
    const int mLocBase = (lane >> 4) * 8;
    const int nLoc     = lane & 15;
    #pragma unroll
    for (int tm = 0; tm < 4; ++tm)
        #pragma unroll
        for (int tn = 0; tn < 2; ++tn) {
            const int nG = colBase + wn * 32 + tn * 16 + nLoc;
            float sc = 0.f, sh = 0.f;
            if (MODE == 1) { sc = scale[nG]; sh = shift[nG]; }
            #pragma unroll
            for (int v = 0; v < 8; ++v) {
                const int mG = rowBase + wm * 64 + tm * 16 + mLocBase + v;
                float val = acc[tm][tn][v];
                if (MODE == 1) {
                    float y = val * sc + sh;
                    if (nG < HH)
                        Zb[(size_t)mG * HH + nG] =
                            __builtin_amdgcn_rcpf(1.0f + __expf(-y));
                    else
                        Cb[(size_t)mG * HH + (nG - HH)] = fmaxf(y, 0.0f);
                } else {
                    Out[(size_t)mG * ldo + nG] = val;
                }
            }
        }
}

// ---------------- sequential time scan ----------------
// one thread per (b,h); h = (1-z)*c + z*h over t; write bf16 next-layer x
__global__ __launch_bounds__(256) void scan_kernel(const float* __restrict__ Zb,
                                                   const float* __restrict__ Cb,
                                                   unsigned short* __restrict__ Xout) {
    int idx = blockIdx.x * 256 + threadIdx.x;          // over BB*HH
    if (idx >= BB * HH) return;
    int b = idx / HH, h = idx - b * HH;
    float hs = 0.0f;
    for (int t = 0; t < TT; ++t) {
        size_t m = (size_t)t * BB + b;
        float z = Zb[m * HH + h];
        float c = Cb[m * HH + h];
        hs = (1.0f - z) * c + z * hs;
        Xout[m * XLD + h] = f2bf(hs);
    }
}

// ---------------- BN + log_softmax over O ----------------
__global__ __launch_bounds__(256) void bn_logsoftmax_kernel(const float* __restrict__ F,
                                                            const float* __restrict__ scale,
                                                            const float* __restrict__ shift,
                                                            float* __restrict__ out) {
    const int row = blockIdx.x;
    const float* f = F + (size_t)row * NPADF;
    __shared__ float red[256];
    float mx = -1e30f;
    for (int o = threadIdx.x; o < OO; o += 256)
        mx = fmaxf(mx, f[o] * scale[o] + shift[o]);
    red[threadIdx.x] = mx;
    __syncthreads();
    for (int s = 128; s > 0; s >>= 1) {
        if (threadIdx.x < s) red[threadIdx.x] = fmaxf(red[threadIdx.x], red[threadIdx.x + s]);
        __syncthreads();
    }
    mx = red[0];
    __syncthreads();
    float sum = 0.0f;
    for (int o = threadIdx.x; o < OO; o += 256)
        sum += __expf(f[o] * scale[o] + shift[o] - mx);
    red[threadIdx.x] = sum;
    __syncthreads();
    for (int s = 128; s > 0; s >>= 1) {
        if (threadIdx.x < s) red[threadIdx.x] += red[threadIdx.x + s];
        __syncthreads();
    }
    const float lse = mx + __logf(red[0]);
    for (int o = threadIdx.x; o < OO; o += 256)
        out[(size_t)row * OO + o] = f[o] * scale[o] + shift[o] - lse;
}

// ---------------- launcher ----------------
extern "C" void kernel_launch(void* const* d_in, const int* in_sizes, int n_in,
                              void* d_out, int out_size, void* d_ws, size_t ws_size,
                              hipStream_t stream) {
    const float* xs    = (const float*)d_in[0];
    const float* w0    = (const float*)d_in[1];
    const float* wr    = (const float*)d_in[2];
    const float* bnz_g = (const float*)d_in[3];
    const float* bnz_b = (const float*)d_in[4];
    const float* bnz_m = (const float*)d_in[5];
    const float* bnz_v = (const float*)d_in[6];
    const float* bnc_g = (const float*)d_in[7];
    const float* bnc_b = (const float*)d_in[8];
    const float* bnc_m = (const float*)d_in[9];
    const float* bnc_v = (const float*)d_in[10];
    const float* wf    = (const float*)d_in[11];
    const float* bnf_g = (const float*)d_in[12];
    const float* bnf_b = (const float*)d_in[13];
    const float* bnf_m = (const float*)d_in[14];
    const float* bnf_v = (const float*)d_in[15];

    // workspace layout
    char* ws = (char*)d_ws;
    unsigned short* xbf = (unsigned short*)ws;                       // MM*XLD bf16 = 52,428,800 B
    size_t off = (size_t)MM * XLD * 2;
    unsigned short* wbf = (unsigned short*)(ws + off);               // NN*XLD bf16 = 4,194,304 B
    off += (size_t)NN * XLD * 2;
    float* scale = (float*)(ws + off);                               // 2048 f32
    float* shift = scale + NN;                                       // 2048 f32
    off += (size_t)2 * NN * 4;
    off = (off + 255) & ~(size_t)255;
    float* zc = (float*)(ws + off);                                  // MM*NN f32 = 209,715,200 B
    float* Zb = zc;
    float* Cb = zc + (size_t)MM * HH;
    float* Fb = zc;                                                  // reuse as [MM][NPADF] logits

    // 1) x -> bf16 (padded K)
    {
        int n = MM * KPAD0;
        convert_x_kernel<<<(n + 255) / 256, 256, 0, stream>>>(xs, xbf);
    }

    // 2) recurrent layers
    for (int layer = 0; layer < LL; ++layer) {
        const int K    = (layer == 0) ? DD : HH;
        const int Kpad = (layer == 0) ? KPAD0 : HH;
        const float* W = (layer == 0) ? w0 : (wr + (size_t)(layer - 1) * NN * HH);

        int wtot = NN * Kpad;
        convert_w_kernel<<<(wtot + 255) / 256, 256, 0, stream>>>(W, wbf, NN, K, Kpad, wtot);

        bnprep_kernel<<<(HH + 255) / 256, 256, 0, stream>>>(
            bnz_g + (size_t)layer * HH, bnz_b + (size_t)layer * HH,
            bnz_m + (size_t)layer * HH, bnz_v + (size_t)layer * HH, scale, shift, HH);
        bnprep_kernel<<<(HH + 255) / 256, 256, 0, stream>>>(
            bnc_g + (size_t)layer * HH, bnc_b + (size_t)layer * HH,
            bnc_m + (size_t)layer * HH, bnc_v + (size_t)layer * HH, scale + HH, shift + HH, HH);

        const int nBlocksN = NN / 128;                 // 16
        const int grid = (MM / 128) * nBlocksN;        // 200 * 16 = 3200
        gemm_wmma_bf16<1><<<grid, 256, 0, stream>>>(xbf, XLD, Kpad, wbf, Kpad,
                                                    scale, shift, Zb, Cb,
                                                    nullptr, 0, nBlocksN);

        scan_kernel<<<(BB * HH + 255) / 256, 256, 0, stream>>>(Zb, Cb, xbf);
    }

    // 3) classifier GEMM (wf padded to 2048 rows, zero-filled)
    {
        int wtot = NPADF * HH;
        convert_w_kernel<<<(wtot + 255) / 256, 256, 0, stream>>>(wf, wbf, OO, HH, HH, wtot);
        bnprep_kernel<<<(OO + 255) / 256, 256, 0, stream>>>(bnf_g, bnf_b, bnf_m, bnf_v,
                                                            scale, shift, OO);
        const int nBlocksN = NPADF / 128;              // 16
        const int grid = (MM / 128) * nBlocksN;        // 3200
        gemm_wmma_bf16<0><<<grid, 256, 0, stream>>>(xbf, XLD, HH, wbf, HH,
                                                    nullptr, nullptr, nullptr, nullptr,
                                                    Fb, NPADF, nBlocksN);
    }

    // 4) BN + log_softmax
    bn_logsoftmax_kernel<<<MM, 256, 0, stream>>>(Fb, scale, shift, (float*)d_out);
}
